// SwitchTransformerMoE_16544214024863
// MI455X (gfx1250) — matrix operand order repaired and checked by reading, not compiled, not run on hardware
//
#include <hip/hip_runtime.h>

typedef __attribute__((ext_vector_type(16))) _Float16 v16h;
typedef __attribute__((ext_vector_type(8)))  _Float16 v8h;
typedef __attribute__((ext_vector_type(8)))  float    v8f;
typedef __attribute__((ext_vector_type(4)))  float    v4f;
typedef __attribute__((ext_vector_type(4)))  int      v4i;

// Problem shape (fixed by the harness reference): B=8, S=8192, D=512, E=209
constexpr int D_DIM = 512;
constexpr int E_DIM = 209;
constexpr int ET    = 14;              // ceil(209/16) expert tiles
constexpr int EPAD  = ET * 16;         // 224 (padded experts, zero-filled)
constexpr int NTOK  = 8 * 8192;        // 65536 tokens
constexpr int WPB   = 4;               // waves per block
constexpr int BLOCK = WPB * 32;        // 128 threads

// LDS slab: 224 rows x 32 halves, row padded 64B->80B (20 dwords) so the
// 16-lane ds_load_b128 stripe (20-dword stride) covers all 64 banks.
constexpr int SLAB_ROW_H = 40;                 // halves per row (80 B)
constexpr int SLAB_H     = EPAD * SLAB_ROW_H;  // halves per slab buffer
constexpr int SEGS       = EPAD * 4;           // 16B segments per slab = 896
constexpr int SEG_ITERS  = SEGS / BLOCK;       // 7 async issues per thread/wave

#if defined(__AMDGCN__) && \
    __has_builtin(__builtin_amdgcn_global_load_async_to_lds_b128) && \
    __has_builtin(__builtin_amdgcn_s_wait_asynccnt)
#define USE_ASYNC 1
#else
#define USE_ASYNC 0
#endif

// Stage one 224x32 f16 slab of gate_w (pre-converted) into LDS.
__device__ __forceinline__ void stage_slab(const _Float16* __restrict__ gwh,
                                           _Float16* slab_dst, int kb, int tid) {
#pragma unroll
  for (int it = 0; it < SEG_ITERS; ++it) {
    const int s   = tid + it * BLOCK;        // 0..895
    const int row = s >> 2;
    const int c   = s & 3;
    const _Float16* src = gwh + (size_t)row * D_DIM + kb + c * 8;
    _Float16*       dst = slab_dst + row * SLAB_ROW_H + c * 8;
#if USE_ASYNC
    __builtin_amdgcn_global_load_async_to_lds_b128(
        (__attribute__((address_space(1))) v4i*)src,
        (__attribute__((address_space(3))) v4i*)dst, 0, 0);
#else
    *(v8h*)dst = *(const v8h*)src;
#endif
  }
}

// Pre-convert gate_w [209,512] fp32 -> [224,512] f16 (padded rows zero).
__global__ __launch_bounds__(256)
void convert_gw_kernel(const float* __restrict__ gw, _Float16* __restrict__ gwh) {
  const int e = blockIdx.x;                  // 0..223
  const size_t ro = (size_t)e * D_DIM;
  for (int k = threadIdx.x; k < D_DIM; k += 256) {
    const float v = (e < E_DIM) ? gw[ro + k] : 0.f;
    gwh[ro + k] = (_Float16)v;
  }
}

__global__ __launch_bounds__(BLOCK)
void moe_gate_kernel(const float* __restrict__ x,
                     const _Float16* __restrict__ gwh,
                     float* __restrict__ out_xflat,
                     float* __restrict__ out_scores,
                     float* __restrict__ out_idx,
                     float* __restrict__ g_hist,
                     float* __restrict__ g_colsum) {
  __shared__ _Float16 slab[2 * SLAB_H];      // 2 x 17.5 KB double buffer
  __shared__ float lds_hist[EPAD];
  __shared__ float lds_col[EPAD];
  const int tid = threadIdx.x;
  for (int i = tid; i < EPAD; i += BLOCK) { lds_hist[i] = 0.f; lds_col[i] = 0.f; }

  const int wave = tid >> 5;
  const int lane = tid & 31;
  const int half = lane >> 4;                // which 16-lane half
  const int ln   = lane & 15;
  const int t0   = (blockIdx.x * WPB + wave) * 16;  // first token of wave tile

  v8f zero = {};
  v8f acc[ET];
#pragma unroll
  for (int et = 0; et < ET; ++et) acc[et] = zero;

  const float* xrow = x + (size_t)(t0 + ln) * D_DIM;
  float*       frow = out_xflat + (size_t)(t0 + ln) * D_DIM;

  stage_slab(gwh, slab, 0, tid);             // prologue: buffer 0
  int cb = 0;
  for (int kb = 0; kb < D_DIM; kb += 32) {
    // ---- A raw global loads issued BEFORE the barriers: their latency
    //      overlaps barrier wait + next-slab staging. ----
    const int ca  = kb + 8 * half;           // K 0..7 (h=0) / 8..15 (h=1)
    const int cbk = kb + 16 + 8 * half;      // K 16..23 / 24..31
    const v4f c0lo = *(const v4f*)(xrow + ca);
    const v4f c0hi = *(const v4f*)(xrow + ca + 4);
    const v4f c1lo = *(const v4f*)(xrow + cbk);
    const v4f c1hi = *(const v4f*)(xrow + cbk + 4);

    __syncthreads();   // all waves finished reading buffer cb^1 (previous iteration)
    if (kb + 32 < D_DIM) {
      stage_slab(gwh, slab + (cb ^ 1) * SLAB_H, kb + 32, tid);
#if USE_ASYNC
      __builtin_amdgcn_s_wait_asynccnt(SEG_ITERS);  // current slab (kb) done
#endif
    }
#if USE_ASYNC
    else {
      __builtin_amdgcn_s_wait_asynccnt(0);
    }
#endif
    __syncthreads();   // slab(kb) visible to all waves

    // ---- A fragment conversion (f16 16x32 layout) + fused NT x_flat copy ----
    v16h a;
    a[0]  = (_Float16)c0lo.x; a[1]  = (_Float16)c0lo.y;
    a[2]  = (_Float16)c0lo.z; a[3]  = (_Float16)c0lo.w;
    a[4]  = (_Float16)c0hi.x; a[5]  = (_Float16)c0hi.y;
    a[6]  = (_Float16)c0hi.z; a[7]  = (_Float16)c0hi.w;
    a[8]  = (_Float16)c1lo.x; a[9]  = (_Float16)c1lo.y;
    a[10] = (_Float16)c1lo.z; a[11] = (_Float16)c1lo.w;
    a[12] = (_Float16)c1hi.x; a[13] = (_Float16)c1hi.y;
    a[14] = (_Float16)c1hi.z; a[15] = (_Float16)c1hi.w;
    __builtin_nontemporal_store(c0lo, (v4f*)(frow + ca));
    __builtin_nontemporal_store(c0hi, (v4f*)(frow + ca + 4));
    __builtin_nontemporal_store(c1lo, (v4f*)(frow + cbk));
    __builtin_nontemporal_store(c1hi, (v4f*)(frow + cbk + 4));

    // ---- 14 expert tiles: B fragments from LDS, software-pipelined with an
    //      explicit 2-entry register double buffer so ds_load latency hides
    //      under the previous WMMA. ----
    union BU { v16h v; v8h h[2]; };
    const _Float16* sl = slab + cb * SLAB_H + (size_t)ln * SLAB_ROW_H + half * 8;
    auto loadB = [&](BU& b, int et) {
      const _Float16* bp = sl + et * (16 * SLAB_ROW_H);
      b.h[0] = *(const v8h*)(bp);
      b.h[1] = *(const v8h*)(bp + 16);
    };
    BU bb[2];
    loadB(bb[0], 0);
#pragma unroll
    for (int et = 0; et < ET; ++et) {
      if (et + 1 < ET) loadB(bb[(et + 1) & 1], et + 1);
      acc[et] = __builtin_amdgcn_wmma_f32_16x16x32_f16(
          false, a, false, bb[et & 1].v, (short)0, acc[et], false, false);
    }
    cb ^= 1;
  }

  // ---- per-token softmax / max / argmax; two-level reductions ----
  // acc[et][j] @ lane L: token = t0 + j + 8*half, expert = et*16 + (L&15)
  float psum[ET];
#pragma unroll
  for (int et = 0; et < ET; ++et) psum[et] = 0.f;

#pragma unroll
  for (int j = 0; j < 8; ++j) {
    float m = -3.4e38f;
    int   im = 0;
#pragma unroll
    for (int et = 0; et < ET; ++et) {
      const int e = et * 16 + ln;
      const float v = acc[et][j];
      if (e < E_DIM && v > m) { m = v; im = e; }   // ascending e: keeps first max
    }
#pragma unroll
    for (int s = 8; s >= 1; s >>= 1) {             // in-half butterfly
      const float om = __shfl_xor(m, s, 32);
      const int   oi = __shfl_xor(im, s, 32);
      if (om > m || (om == m && oi < im)) { m = om; im = oi; }
    }
    float ssum = 0.f;
#pragma unroll
    for (int et = 0; et < ET; ++et) {
      const int e = et * 16 + ln;
      if (e < E_DIM) ssum += __expf(acc[et][j] - m);
    }
#pragma unroll
    for (int s = 8; s >= 1; s >>= 1) ssum += __shfl_xor(ssum, s, 32);
    const float inv = 1.0f / ssum;                 // max prob = e^0 / ssum

    if (ln == 0) {
      const int tok = t0 + j + 8 * half;
      out_scores[tok] = inv;
      out_idx[tok]    = (float)im;
      atomicAdd(&lds_hist[im], 1.0f);
    }
#pragma unroll
    for (int et = 0; et < ET; ++et) {
      const int e = et * 16 + ln;
      if (e < E_DIM) psum[et] += __expf(acc[et][j] - m) * inv;
    }
  }
#pragma unroll
  for (int et = 0; et < ET; ++et) {
    const int e = et * 16 + ln;
    if (e < E_DIM) atomicAdd(&lds_col[e], psum[et]);
  }
  __syncthreads();
  for (int i = tid; i < E_DIM; i += BLOCK) {
    atomicAdd(&g_hist[i],   lds_hist[i]);
    atomicAdd(&g_colsum[i], lds_col[i]);
  }
}

__global__ __launch_bounds__(256)
void moe_finalize_kernel(const float* __restrict__ ec_in,
                         const float* __restrict__ gps_in,
                         const float* __restrict__ g_hist,
                         const float* __restrict__ g_colsum,
                         float* __restrict__ out_loss,
                         float* __restrict__ out_counts,
                         float* __restrict__ out_gps) {
  __shared__ float red[256];
  const int tid = threadIdx.x;
  float part = 0.f;
  for (int e = tid; e < E_DIM; e += 256) {
    const float h = g_hist[e];
    const float c = g_colsum[e];
    out_counts[e] = 0.9f * ec_in[e]  + 0.1f * h;
    out_gps[e]    = 0.9f * gps_in[e] + 0.1f * c;
    part += h * c;
  }
  red[tid] = part;
  __syncthreads();
  for (int s = 128; s > 0; s >>= 1) {
    if (tid < s) red[tid] += red[tid + s];
    __syncthreads();
  }
  if (tid == 0) {
    *out_loss = (float)E_DIM * red[0] / ((float)NTOK * (float)NTOK);
  }
}

extern "C" void kernel_launch(void* const* d_in, const int* in_sizes, int n_in,
                              void* d_out, int out_size, void* d_ws, size_t ws_size,
                              hipStream_t stream) {
  const float* x   = (const float*)d_in[0];
  const float* gw  = (const float*)d_in[1];
  const float* ec  = (const float*)d_in[2];
  const float* gps = (const float*)d_in[3];

  // Output tuple, concatenated flat:
  // x_flat[N*D] | gate_scores[N] | expert_indices[N] | loss[1] | counts[E] | gps[E]
  float* out        = (float*)d_out;
  float* out_xflat  = out;
  float* out_scores = out_xflat + (size_t)NTOK * D_DIM;
  float* out_idx    = out_scores + NTOK;
  float* out_loss   = out_idx + NTOK;
  float* out_counts = out_loss + 1;
  float* out_gps    = out_counts + E_DIM;

  // Workspace: g_hist[224] | g_colsum[224] | gwh f16 [224*512]
  float*    g_hist   = (float*)d_ws;
  float*    g_colsum = g_hist + EPAD;
  _Float16* gwh      = (_Float16*)((char*)d_ws + 2 * EPAD * sizeof(float));

  (void)hipMemsetAsync(d_ws, 0, 2 * EPAD * sizeof(float), stream);
  convert_gw_kernel<<<EPAD, 256, 0, stream>>>(gw, gwh);

  const int tiles = NTOK / 16;               // 4096
  dim3 grid(tiles / WPB);                    // 1024 blocks, 4 waves each
  moe_gate_kernel<<<grid, BLOCK, 0, stream>>>(x, gwh, out_xflat, out_scores,
                                              out_idx, g_hist, g_colsum);
  moe_finalize_kernel<<<1, 256, 0, stream>>>(ec, gps, g_hist, g_colsum,
                                             out_loss, out_counts, out_gps);
}
